// InjectionMultiHeadSelfAttention_31550829756927
// MI455X (gfx1250) — matrix operand order repaired and checked by reading, not compiled, not run on hardware
//
#include <hip/hip_runtime.h>

#define DEV __device__ __forceinline__

typedef __bf16 bf16_t;
typedef __attribute__((ext_vector_type(8)))  bf16_t v8bf;
typedef __attribute__((ext_vector_type(16))) bf16_t v16bf;
typedef __attribute__((ext_vector_type(8)))  float  v8f;
typedef __attribute__((ext_vector_type(4)))  float  v4f;
typedef __attribute__((ext_vector_type(4)))  int    v4i;

static constexpr int B_  = 8;
static constexpr int N_  = 1024;
static constexpr int D_  = 768;
static constexpr int H_  = 12;
static constexpr int HD_ = 64;
static constexpr int BN_ = B_ * N_;       // 8192
static constexpr int DT_ = D_ / 16;       // 48 column tiles
static constexpr float INV_SCALE = 8.0f;  // reference divides by HD^-0.5 => *sqrt(64)

// -------- CDNA5 async global->LDS path (ASYNCcnt), with compile-safe fallback --------
#if defined(__has_builtin)
#if __has_builtin(__builtin_amdgcn_global_load_async_to_lds_b128)
#define HAVE_ASYNC_LDS 1
#endif
#endif
#ifndef HAVE_ASYNC_LDS
#define HAVE_ASYNC_LDS 0
#endif

DEV void async_copy16(const bf16_t* g, bf16_t* l) {
#if HAVE_ASYNC_LDS
  __builtin_amdgcn_global_load_async_to_lds_b128(
      (__attribute__((address_space(1))) v4i*)(g),
      (__attribute__((address_space(3))) v4i*)(l), 0, 0);
#else
  *(v8bf*)l = *(const v8bf*)g;  // sync fallback: global load + ds_store
#endif
}

template <int Nw>
DEV void wait_async() {
#if HAVE_ASYNC_LDS
#if __has_builtin(__builtin_amdgcn_s_wait_asynccnt)
  __builtin_amdgcn_s_wait_asynccnt(Nw);
#else
  asm volatile("s_wait_asynccnt %0" ::"i"(Nw) : "memory");
#endif
#endif
}

// ---------------- fragment helpers (bf16 16x16x32 WMMA, wave32) ----------------
DEV v16bf frag_cat(v8bf lo, v8bf hi) {
  v16bf r;
#pragma unroll
  for (int i = 0; i < 8; ++i) { r[i] = lo[i]; r[i + 8] = hi[i]; }
  return r;
}

// A 16x32: lane<16 -> row=lane, K {0..7,16..23}; lane>=16 -> row=lane-16, K {8..15,24..31}
DEV v16bf load_frag_a(const bf16_t* p, int ld, int row0, int k0) {
  const int lane = threadIdx.x & 31;
  const int half = lane >> 4;
  const bf16_t* q = p + (size_t)(row0 + (lane & 15)) * ld + k0 + half * 8;
  v8bf lo = *(const v8bf*)(q);
  v8bf hi = *(const v8bf*)(q + 16);
  return frag_cat(lo, hi);
}

// B 32x16: lane = col n; lanes<16 hold K {0..7,16..23}, lanes>=16 hold K {8..15,24..31}
DEV v16bf load_frag_b(const bf16_t* p, int ld, int n0, int k0) {
  const int lane = threadIdx.x & 31;
  const int half = lane >> 4;
  const bf16_t* q = p + (size_t)(n0 + (lane & 15)) * ld + k0 + half * 8;
  v8bf lo = *(const v8bf*)(q);
  v8bf hi = *(const v8bf*)(q + 16);
  return frag_cat(lo, hi);
}

DEV v8f wmma_bf16(v16bf a, v16bf b, v8f c) {
  return __builtin_amdgcn_wmma_f32_16x16x32_bf16(false, a, false, b, (short)0, c,
                                                 false, false);
}

// ---------------- K1: f32 -> bf16 conversion ----------------
__global__ void cvt_kernel(const float* __restrict__ s, bf16_t* __restrict__ d, int n) {
  int i = blockIdx.x * 256 + threadIdx.x;
  if (i < n) d[i] = (bf16_t)s[i];
}

__global__ void cvt_add_kernel(const float* __restrict__ s0, const float* __restrict__ s1,
                               bf16_t* __restrict__ d, int n) {
  int i = blockIdx.x * 256 + threadIdx.x;
  if (i < n) d[i] = (bf16_t)(s0[i] + s1[i]);
}

// ---------------- K2: fused projections ----------------
// qs = x@(Wq+Wqt)^T+(bq+bqt), qt = x@Wqt^T+bqt, k = x@Wk^T+bk, v = x@Wv^T+bv.
// Tile remap: 4 waves of a block share dt (identical weight fragments -> WGP$ dedupe),
// consecutive mtiles.  v stored transposed [B,H,HD,N] for contiguous B-frags in attn.
__global__ __launch_bounds__(128) void proj_qkv_kernel(
    const bf16_t* __restrict__ xb,
    const bf16_t* __restrict__ wqs, const bf16_t* __restrict__ wqt,
    const bf16_t* __restrict__ wk,  const bf16_t* __restrict__ wv,
    const float* __restrict__ b_q,  const float* __restrict__ b_qt,
    const float* __restrict__ b_k,  const float* __restrict__ b_v,
    bf16_t* __restrict__ qs, bf16_t* __restrict__ qt,
    bf16_t* __restrict__ kk, bf16_t* __restrict__ vt) {
  __shared__ __align__(16) bf16_t tlds[4][256];

  const int lane  = threadIdx.x & 31;
  const int wave  = threadIdx.x >> 5;
  const int t     = blockIdx.x * 4 + wave;
  const int dt    = t >> 9;             // 0..47 : shared by the block's 4 waves
  const int mtile = t & 511;            // 0..511
  const int b     = mtile >> 6;
  const int n0    = (mtile & 63) << 4;
  const int d0    = dt << 4;
  const int h     = d0 >> 6;
  const int hd0   = d0 & 63;
  const int bh    = b * H_ + h;
  const int gm0   = mtile << 4;
  const int half  = lane >> 4;
  const int nlane = lane & 15;

  v8f cqs = {}, cqt = {}, ck = {}, cv = {};
  {
    const float vb_qs = b_q[d0 + nlane] + b_qt[d0 + nlane];
    const float vb_qt = b_qt[d0 + nlane];
    const float vb_k  = b_k[d0 + nlane];
    const float vb_v  = b_v[d0 + nlane];
#pragma unroll
    for (int i = 0; i < 8; ++i) { cqs[i] = vb_qs; cqt[i] = vb_qt; ck[i] = vb_k; cv[i] = vb_v; }
  }

  for (int k0 = 0; k0 < D_; k0 += 32) {
    v16bf a  = load_frag_a(xb, D_, gm0, k0);
    v16bf fq = load_frag_b(wqs, D_, d0, k0);
    v16bf ft = load_frag_b(wqt, D_, d0, k0);
    v16bf fk = load_frag_b(wk,  D_, d0, k0);
    v16bf fv = load_frag_b(wv,  D_, d0, k0);
    cqs = wmma_bf16(a, fq, cqs);
    cqt = wmma_bf16(a, ft, cqt);
    ck  = wmma_bf16(a, fk, ck);
    cv  = wmma_bf16(a, fv, cv);
  }

  // v: transposed store, contiguous along n
  {
    v8bf pk;
#pragma unroll
    for (int r = 0; r < 8; ++r) pk[r] = (bf16_t)cv[r];
    bf16_t* dst = vt + (size_t)(bh * HD_ + hd0 + nlane) * N_ + n0 + half * 8;
    *(v8bf*)dst = pk;
  }

  // qs/qt/k: per-wave LDS transpose -> contiguous b128 stores
  bf16_t* l = tlds[wave];
  auto store_rowmajor = [&](v8f c, bf16_t* base) {
#pragma unroll
    for (int r = 0; r < 8; ++r) l[(r + half * 8) * 16 + nlane] = (bf16_t)c[r];
    asm volatile("s_wait_dscnt 0x0" ::: "memory");
    v8bf v = *(const v8bf*)(l + (lane >> 1) * 16 + (lane & 1) * 8);
    bf16_t* dst = base + ((size_t)bh * N_ + n0 + (lane >> 1)) * HD_ + hd0 + (lane & 1) * 8;
    *(v8bf*)dst = v;
    asm volatile("s_wait_dscnt 0x0" ::: "memory");
  };
  store_rowmajor(cqs, qs);
  store_rowmajor(cqt, qt);
  store_rowmajor(ck,  kk);
}

// ---------------- K3: attention + corr, block-cooperative flash ----------------
// 4 waves/block share one (b,h); K/V 32-column tiles double-buffered in LDS via the
// CDNA5 async-to-LDS path (ASYNCcnt).  corr written as NT 128B-coalesced stores.
__global__ __launch_bounds__(128) void attn_kernel(
    const bf16_t* __restrict__ qs, const bf16_t* __restrict__ qt,
    const bf16_t* __restrict__ kk, const bf16_t* __restrict__ vt,
    bf16_t* __restrict__ ctx, float* __restrict__ corr) {
  __shared__ __align__(16) bf16_t Kbuf[2][32 * 64];   // [m rows][64 d]
  __shared__ __align__(16) bf16_t Vbuf[2][64 * 32];   // [hd rows][32 m]
  __shared__ __align__(16) bf16_t plds[4][16 * 32];   // per-wave P repack
  __shared__ __align__(16) float  cbuf[4][16 * 32];   // per-wave corr transpose

  const int tid   = threadIdx.x;
  const int lane  = tid & 31;
  const int wave  = tid >> 5;
  const int bh    = blockIdx.x >> 4;                  // 16 blocks per (b,h)
  const int q16   = ((blockIdx.x & 15) << 2) + wave;  // query tile 0..63
  const int b     = bh / H_;
  const int h     = bh - b * H_;
  const int half  = lane >> 4;
  const int nlane = lane & 15;

  const bf16_t* qs_bh = qs + (size_t)bh * N_ * HD_;
  const bf16_t* qt_bh = qt + (size_t)bh * N_ * HD_;
  const bf16_t* kk_bh = kk + (size_t)bh * N_ * HD_;
  const bf16_t* vt_bh = vt + (size_t)bh * HD_ * N_;
  float* corr_b = corr + (size_t)bh * N_ * N_ + (size_t)(q16 * 16) * N_;

  // stage one 32-column K/V tile (4KB each) -> per thread 2x16B for K, 2x16B for V
  auto stage = [&](int mt, int buf) {
    const int c0 = mt * 32;
    const int e0 = tid * 16;                 // flat element offset (tile row-major)
    async_copy16(kk_bh + (size_t)c0 * HD_ + e0,     Kbuf[buf] + e0);
    async_copy16(kk_bh + (size_t)c0 * HD_ + e0 + 8, Kbuf[buf] + e0 + 8);
    const int r0 = e0 >> 5;                  // V: hd row (32 elems per row)
    const int cA = e0 & 31;
    async_copy16(vt_bh + (size_t)r0 * N_ + c0 + cA,     Vbuf[buf] + e0);
    async_copy16(vt_bh + (size_t)r0 * N_ + c0 + cA + 8, Vbuf[buf] + e0 + 8);
  };

  const v16bf a_qs0 = load_frag_a(qs_bh, HD_, q16 * 16, 0);
  const v16bf a_qs1 = load_frag_a(qs_bh, HD_, q16 * 16, 32);
  const v16bf a_qt0 = load_frag_a(qt_bh, HD_, q16 * 16, 0);
  const v16bf a_qt1 = load_frag_a(qt_bh, HD_, q16 * 16, 32);

  stage(0, 0);
  stage(1, 1);

  v8f o0 = {}, o1 = {}, o2 = {}, o3 = {};
  v8f mrun, lrun;
#pragma unroll
  for (int i = 0; i < 8; ++i) { mrun[i] = -3.0e38f; lrun[i] = 0.0f; }

  bf16_t* l  = plds[wave];
  float*  cw = cbuf[wave];

  for (int mt = 0; mt < N_ / 32; ++mt) {
    const int cur = mt & 1;
    const int c0  = mt * 32;
    if (mt < 31) wait_async<4>(); else wait_async<0>();  // current tile landed
    __syncthreads();                                      // ...for every wave

    v16bf bk0 = load_frag_b(Kbuf[cur], HD_, 0, 0);
    v16bf bk1 = load_frag_b(Kbuf[cur], HD_, 0, 32);
    v16bf bk2 = load_frag_b(Kbuf[cur], HD_, 16, 0);
    v16bf bk3 = load_frag_b(Kbuf[cur], HD_, 16, 32);

    v8f cc0 = {}, cc1 = {}, ss0 = {}, ss1 = {};
    cc0 = wmma_bf16(a_qt0, bk0, cc0); cc0 = wmma_bf16(a_qt1, bk1, cc0);
    ss0 = wmma_bf16(a_qs0, bk0, ss0); ss0 = wmma_bf16(a_qs1, bk1, ss0);
    cc1 = wmma_bf16(a_qt0, bk2, cc1); cc1 = wmma_bf16(a_qt1, bk3, cc1);
    ss1 = wmma_bf16(a_qs0, bk2, ss1); ss1 = wmma_bf16(a_qs1, bk3, ss1);

#pragma unroll
    for (int r = 0; r < 8; ++r) {
      cc0[r] *= INV_SCALE; cc1[r] *= INV_SCALE;
      ss0[r] *= INV_SCALE; ss1[r] *= INV_SCALE;  // qs = q+qt -> ss already includes corr
    }

    // corr: LDS transpose -> 128B-coalesced non-temporal b128 stores (402MB stream)
#pragma unroll
    for (int r = 0; r < 8; ++r) {
      cw[(r + 8 * half) * 32 + nlane]      = cc0[r];
      cw[(r + 8 * half) * 32 + 16 + nlane] = cc1[r];
    }
    asm volatile("s_wait_dscnt 0x0" ::: "memory");
#pragma unroll
    for (int j = 0; j < 4; ++j) {
      const int chunk = j * 32 + lane;
      const int rr = chunk >> 3;
      const int c4 = (chunk & 7) * 4;
      v4f val = *(const v4f*)(cw + rr * 32 + c4);
      __builtin_nontemporal_store(val, (v4f*)(corr_b + (size_t)rr * N_ + c0 + c4));
    }

    // online softmax update + repack P as bf16 A-fragment via LDS
#pragma unroll
    for (int r = 0; r < 8; ++r) {
      float mo = mrun[r];
      float mx = fmaxf(ss0[r], ss1[r]);
      mx = fmaxf(mx, __shfl_xor(mx, 1, 32));
      mx = fmaxf(mx, __shfl_xor(mx, 2, 32));
      mx = fmaxf(mx, __shfl_xor(mx, 4, 32));
      mx = fmaxf(mx, __shfl_xor(mx, 8, 32));
      float mn = fmaxf(mo, mx);
      float p0 = __expf(ss0[r] - mn);
      float p1 = __expf(ss1[r] - mn);
      float rs = p0 + p1;
      rs += __shfl_xor(rs, 1, 32);
      rs += __shfl_xor(rs, 2, 32);
      rs += __shfl_xor(rs, 4, 32);
      rs += __shfl_xor(rs, 8, 32);
      float sf = __expf(mo - mn);
      mrun[r] = mn;
      lrun[r] = lrun[r] * sf + rs;
      o0[r] *= sf; o1[r] *= sf; o2[r] *= sf; o3[r] *= sf;
      l[(r + 8 * half) * 32 + nlane]      = (bf16_t)p0;
      l[(r + 8 * half) * 32 + 16 + nlane] = (bf16_t)p1;
    }
    asm volatile("s_wait_dscnt 0x0" ::: "memory");

    v16bf ap  = load_frag_a(l, 32, 0, 0);
    v16bf bv0 = load_frag_b(Vbuf[cur], 32, 0,  0);
    v16bf bv1 = load_frag_b(Vbuf[cur], 32, 16, 0);
    v16bf bv2 = load_frag_b(Vbuf[cur], 32, 32, 0);
    v16bf bv3 = load_frag_b(Vbuf[cur], 32, 48, 0);
    o0 = wmma_bf16(ap, bv0, o0);
    o1 = wmma_bf16(ap, bv1, o1);
    o2 = wmma_bf16(ap, bv2, o2);
    o3 = wmma_bf16(ap, bv3, o3);

    __syncthreads();                       // everyone done reading buf[cur]
    if (mt + 2 < N_ / 32) stage(mt + 2, cur);
  }

  // normalize and store context [B,N,D] bf16
  bf16_t* cb = ctx + ((size_t)b * N_ + q16 * 16 + 8 * half) * D_ + h * HD_ + nlane;
#pragma unroll
  for (int r = 0; r < 8; ++r) {
    float inv = 1.0f / lrun[r];
    cb[(size_t)r * D_]      = (bf16_t)(o0[r] * inv);
    cb[(size_t)r * D_ + 16] = (bf16_t)(o1[r] * inv);
    cb[(size_t)r * D_ + 32] = (bf16_t)(o2[r] * inv);
    cb[(size_t)r * D_ + 48] = (bf16_t)(o3[r] * inv);
  }
}

// ---------------- K4: output projection (f32 result, NT stores) ----------------
__global__ __launch_bounds__(128) void proj_out_kernel(
    const bf16_t* __restrict__ ctx, const bf16_t* __restrict__ wp,
    const float* __restrict__ b_p, float* __restrict__ out) {
  const int lane  = threadIdx.x & 31;
  const int wave  = threadIdx.x >> 5;
  const int t     = blockIdx.x * 4 + wave;
  const int dt    = t >> 9;             // block's 4 waves share the weight tile
  const int mtile = t & 511;
  const int gm0   = mtile << 4;
  const int d0    = dt << 4;
  const int half  = lane >> 4;
  const int nlane = lane & 15;

  v8f c = {};
  {
    const float bv = b_p[d0 + nlane];
#pragma unroll
    for (int i = 0; i < 8; ++i) c[i] = bv;
  }
  for (int k0 = 0; k0 < D_; k0 += 32) {
    v16bf a = load_frag_a(ctx, D_, gm0, k0);
    v16bf f = load_frag_b(wp,  D_, d0, k0);
    c = wmma_bf16(a, f, c);
  }
  float* dst = out + (size_t)(gm0 + 8 * half) * D_ + d0 + nlane;
#pragma unroll
  for (int r = 0; r < 8; ++r) __builtin_nontemporal_store(c[r], dst + (size_t)r * D_);
}

// ---------------- host ----------------
extern "C" void kernel_launch(void* const* d_in, const int* in_sizes, int n_in,
                              void* d_out, int out_size, void* d_ws, size_t ws_size,
                              hipStream_t stream) {
  const float* x    = (const float*)d_in[0];
  const float* w_q  = (const float*)d_in[1];
  const float* b_q  = (const float*)d_in[2];
  const float* w_k  = (const float*)d_in[3];
  const float* b_k  = (const float*)d_in[4];
  const float* w_v  = (const float*)d_in[5];
  const float* b_v  = (const float*)d_in[6];
  const float* w_qt = (const float*)d_in[7];
  const float* b_qt = (const float*)d_in[8];
  const float* w_p  = (const float*)d_in[9];
  const float* b_p  = (const float*)d_in[10];

  float* out  = (float*)d_out;                      // [B,N,D]
  float* corr = out + (size_t)B_ * N_ * D_;         // [B,H,N,N]

  char* ws = (char*)d_ws;
  size_t off = 0;
  auto carve = [&](size_t nelem) -> bf16_t* {
    bf16_t* p = (bf16_t*)(ws + off);
    off = (off + nelem * sizeof(bf16_t) + 255) & ~(size_t)255;
    return p;
  };
  const size_t NXD = (size_t)BN_ * D_;   // 6291456
  const size_t NWW = (size_t)D_ * D_;    // 589824
  bf16_t* xb  = carve(NXD);
  bf16_t* wqs = carve(NWW);
  bf16_t* wqt = carve(NWW);
  bf16_t* wk  = carve(NWW);
  bf16_t* wv  = carve(NWW);
  bf16_t* wp  = carve(NWW);
  bf16_t* qsb = carve(NXD);
  bf16_t* qtb = carve(NXD);
  bf16_t* kkb = carve(NXD);
  bf16_t* vtb = carve(NXD);
  bf16_t* ctx = carve(NXD);
  (void)ws_size;

  cvt_kernel<<<(int)((NXD + 255) / 256), 256, 0, stream>>>(x, xb, (int)NXD);
  cvt_add_kernel<<<(int)((NWW + 255) / 256), 256, 0, stream>>>(w_q, w_qt, wqs, (int)NWW);
  cvt_kernel<<<(int)((NWW + 255) / 256), 256, 0, stream>>>(w_qt, wqt, (int)NWW);
  cvt_kernel<<<(int)((NWW + 255) / 256), 256, 0, stream>>>(w_k,  wk,  (int)NWW);
  cvt_kernel<<<(int)((NWW + 255) / 256), 256, 0, stream>>>(w_v,  wv,  (int)NWW);
  cvt_kernel<<<(int)((NWW + 255) / 256), 256, 0, stream>>>(w_p,  wp,  (int)NWW);

  proj_qkv_kernel<<<(BN_ / 16) * DT_ / 4, 128, 0, stream>>>(
      xb, wqs, wqt, wk, wv, b_q, b_qt, b_k, b_v, qsb, qtb, kkb, vtb);

  attn_kernel<<<B_ * H_ * (N_ / 16) / 4, 128, 0, stream>>>(qsb, qtb, kkb, vtb, ctx, corr);

  proj_out_kernel<<<(BN_ / 16) * DT_ / 4, 128, 0, stream>>>(ctx, wp, b_p, out);
}
